// LSTM_T_26594437497127
// MI455X (gfx1250) — matrix operand order repaired and checked by reading, not compiled, not run on hardware
//
#include <hip/hip_runtime.h>
#include <hip/hip_bf16.h>
#include <math.h>

typedef __attribute__((ext_vector_type(16))) __bf16 v16bf;
typedef __attribute__((ext_vector_type(8)))  float  v8f;

#define Hn 1024
#define Ln 512
#define Dn 64
#define Bn 2
#define STR 1032   // padded A row stride in halves: 2064 B, 516 words -> +4 banks/row
#define QPR 129    // uint4 (8-half) quads per padded row

__device__ __forceinline__ unsigned short f2bf(float x) {
    unsigned u = __float_as_uint(x);
    unsigned r = (u + 0x7FFFu + ((u >> 16) & 1u)) >> 16;
    return (unsigned short)r;
}
__device__ __forceinline__ float clip8(float x)  { return fminf(fmaxf(x, -8.0f), 8.0f); }
__device__ __forceinline__ float clip10(float x) { return fminf(fmaxf(x, -10.0f), 10.0f); }
__device__ __forceinline__ float sigm(float x)   { return 1.0f / (1.0f + __expf(-x)); }
__device__ __forceinline__ float gelu(float x)   { return 0.5f * x * (1.0f + erff(x * 0.70710678118654752f)); }

// ---------------- weights fp32 -> bf16 (once, stays in L2: 10 MB) -------------
__global__ __launch_bounds__(256) void w2bf_kernel(const float* __restrict__ Wf,
                                                   const float* __restrict__ Wi,
                                                   const float* __restrict__ Wg,
                                                   const float* __restrict__ Wo,
                                                   const float* __restrict__ Wp,
                                                   unsigned short* __restrict__ out) {
    size_t idx = ((size_t)blockIdx.x * 256 + threadIdx.x) * 4;
    if (idx >= (size_t)5 * Hn * Hn) return;
    int w = (int)(idx >> 20);
    size_t rem = idx & ((1u << 20) - 1);
    const float* src = (w == 0) ? Wf : (w == 1) ? Wi : (w == 2) ? Wg : (w == 3) ? Wo : Wp;
    float4 v = *(const float4*)(src + rem);
    ushort4 o;
    o.x = f2bf(v.x); o.y = f2bf(v.y); o.z = f2bf(v.z); o.w = f2bf(v.w);
    *(ushort4*)(out + idx) = o;
}

// ---------------- fused pF/pI/pO projection: one 256 MB streaming pass --------
__global__ __launch_bounds__(256) void proj_kernel(const float* __restrict__ trans,
                                                   const float* __restrict__ tF,
                                                   const float* __restrict__ tI,
                                                   const float* __restrict__ tO,
                                                   float* __restrict__ pF,
                                                   float* __restrict__ pI,
                                                   float* __restrict__ pO) {
    __shared__ float sF[Ln], sI[Ln], sO[Ln];
    int tid = threadIdx.x;
    for (int l = tid; l < Ln; l += 256) { sF[l] = tF[l]; sI[l] = tI[l]; sO[l] = tO[l]; }
    __syncthreads();

    int bd = blockIdx.x;                       // bd = b*64 + d
    size_t base = (size_t)bd * Ln * Hn + (size_t)tid * 4;
    float aF0 = 0, aF1 = 0, aF2 = 0, aF3 = 0;
    float aI0 = 0, aI1 = 0, aI2 = 0, aI3 = 0;
    float aO0 = 0, aO1 = 0, aO2 = 0, aO3 = 0;
    for (int l = 0; l < Ln; ++l) {
        float4 v = *(const float4*)(trans + base + (size_t)l * Hn);
        float wf = sF[l], wi = sI[l], wo = sO[l];
        aF0 += v.x * wf; aF1 += v.y * wf; aF2 += v.z * wf; aF3 += v.w * wf;
        aI0 += v.x * wi; aI1 += v.y * wi; aI2 += v.z * wi; aI3 += v.w * wi;
        aO0 += v.x * wo; aO1 += v.y * wo; aO2 += v.z * wo; aO3 += v.w * wo;
    }
    size_t ob = (size_t)bd * Hn + (size_t)tid * 4;
    *(float4*)(pF + ob) = make_float4(aF0, aF1, aF2, aF3);
    *(float4*)(pI + ob) = make_float4(aI0, aI1, aI2, aI3);
    *(float4*)(pO + ob) = make_float4(aO0, aO1, aO2, aO3);
}

// ---------------- zero initial state ------------------------------------------
__global__ __launch_bounds__(256) void init_kernel(float* __restrict__ c0, float* __restrict__ h0) {
    int t = threadIdx.x;
    for (int k = t; k < Bn * Hn; k += 256) { c0[k] = 0.0f; h0[k] = 0.0f; }
}

// ---------------- one recurrence step: LN + 5 WMMA matmuls + gates ------------
// grid = 64 blocks (16 h-columns each); block = 256 threads (8 waves, waves 0..4
// own Wf/Wi/Wg/Wo/Wp respectively). A operands kept as fully padded 16-row bf16
// matrices in LDS so every lane's WMMA gather is branch-free and uniform.
__global__ __launch_bounds__(256) void step_kernel(const unsigned short* __restrict__ Wb, // [5][H][H] bf16
                                                   const float* __restrict__ pF,
                                                   const float* __restrict__ pI,
                                                   const float* __restrict__ pO,
                                                   const float* __restrict__ bfv,
                                                   const float* __restrict__ biv,
                                                   const float* __restrict__ bgv,
                                                   const float* __restrict__ bov,
                                                   const float* __restrict__ bpv,
                                                   const float* __restrict__ cg,
                                                   const float* __restrict__ cbt,
                                                   const float* __restrict__ hg,
                                                   const float* __restrict__ hbt,
                                                   const unsigned char* __restrict__ mask,
                                                   const float* __restrict__ c_in,
                                                   const float* __restrict__ h_in,
                                                   float* __restrict__ c_out,
                                                   float* __restrict__ h_out,
                                                   int d) {
    // s_A[0] = padded nH (waves 0..3), s_A[1] = padded nC (wave 4); rows 2..15 zero.
    __shared__ __align__(16) unsigned short s_A[2][16][STR];   // 66 KB
    __shared__ float redw[8][8];
    __shared__ float s_sum[8];
    __shared__ float s_mean[4], s_rstd[4];
    __shared__ float s_z[5][Bn][16];

    int tid  = threadIdx.x;
    int wave = tid >> 5;
    int lane = tid & 31;

    // ---- phase 0: zero padded rows 2..15 of both A matrices (b128 stores) ----
    {
        uint4 z = make_uint4(0, 0, 0, 0);
        for (int idx = tid; idx < 2 * 14 * QPR; idx += 256) {
            int mat = idx / (14 * QPR);
            int rem = idx - mat * (14 * QPR);
            int row = 2 + rem / QPR;
            int qc  = rem - (rem / QPR) * QPR;
            *(uint4*)&s_A[mat][row][qc * 8] = z;
        }
    }

    // ---- phase 1: LN statistics for c[0],c[1],h[0],h[1] (shuffle butterfly) --
    float v0 = 0, v1 = 0, v2 = 0, v3 = 0, v4 = 0, v5 = 0, v6 = 0, v7 = 0;
    for (int k = tid; k < Hn; k += 256) {
        float c0 = c_in[k], c1 = c_in[Hn + k];
        float h0 = h_in[k], h1 = h_in[Hn + k];
        v0 += c0; v1 += c0 * c0;
        v2 += c1; v3 += c1 * c1;
        v4 += h0; v5 += h0 * h0;
        v6 += h1; v7 += h1 * h1;
    }
    for (int off = 16; off > 0; off >>= 1) {
        v0 += __shfl_xor(v0, off); v1 += __shfl_xor(v1, off);
        v2 += __shfl_xor(v2, off); v3 += __shfl_xor(v3, off);
        v4 += __shfl_xor(v4, off); v5 += __shfl_xor(v5, off);
        v6 += __shfl_xor(v6, off); v7 += __shfl_xor(v7, off);
    }
    if (lane == 0) {
        redw[wave][0] = v0; redw[wave][1] = v1; redw[wave][2] = v2; redw[wave][3] = v3;
        redw[wave][4] = v4; redw[wave][5] = v5; redw[wave][6] = v6; redw[wave][7] = v7;
    }
    __syncthreads();
    if (tid < 8) {
        float a = 0.0f;
        for (int w = 0; w < 8; ++w) a += redw[w][tid];
        s_sum[tid] = a;
    }
    __syncthreads();
    if (tid < 4) {
        float mean = s_sum[2 * tid] * (1.0f / Hn);
        float var  = s_sum[2 * tid + 1] * (1.0f / Hn) - mean * mean;
        s_mean[tid] = mean;
        s_rstd[tid] = rsqrtf(var + 1e-5f);
    }
    __syncthreads();

    // ---- phase 2: normalized rows 0,1 -> bf16 in padded LDS A matrices ----
    for (int k = tid; k < Hn; k += 256) {
        float gcv = cg[k], bcv = cbt[k], ghv = hg[k], bhv = hbt[k];
        s_A[1][0][k] = f2bf((c_in[k]      - s_mean[0]) * s_rstd[0] * gcv + bcv);
        s_A[1][1][k] = f2bf((c_in[Hn + k] - s_mean[1]) * s_rstd[1] * gcv + bcv);
        s_A[0][0][k] = f2bf((h_in[k]      - s_mean[2]) * s_rstd[2] * ghv + bhv);
        s_A[0][1][k] = f2bf((h_in[Hn + k] - s_mean[3]) * s_rstd[3] * ghv + bhv);
    }
    __syncthreads();

    // ---- phase 3: five 16x16 output tiles via v_wmma_f32_16x16x32_bf16 ----
    if (wave < 5) {
        const unsigned short* nA = &s_A[(wave == 4) ? 1 : 0][0][0];
        int m  = lane & 15;
        int hi = lane >> 4;                       // lanes 16-31: K offsets 8-15, 24-31
        int col = blockIdx.x * 16 + m;            // output column = row of W
        const unsigned short* wrow = Wb + ((size_t)wave << 20) + (size_t)col * Hn;
        int rowbase = m * STR + (hi ? 8 : 0);     // branch-free lane address

        union VB { v16bf v; uint4 q[2]; } aa, bb;
        v8f acc = {};
        for (int kk = 0; kk < Hn; kk += 32) {
            // A operand (16x32 bf16), uniform two b128 LDS loads per lane
            aa.q[0] = *(const uint4*)(nA + rowbase + kk);        // K = base..base+7
            aa.q[1] = *(const uint4*)(nA + rowbase + kk + 16);   // K = base+16..+23
            // B operand (32x16 bf16): lane n holds W[col][k], K = base..+7, base+16..+23
            const uint4* gp = (const uint4*)(wrow + kk + (hi ? 8 : 0));
            bb.q[0] = gp[0];
            bb.q[1] = gp[2];
            acc = __builtin_amdgcn_wmma_f32_16x16x32_bf16(false, aa.v, false, bb.v,
                                                          (short)0, acc, false, false);
        }
        // C/D layout: lanes 0-15 are columns; VGPR r holds row r (r<8).
        if (lane < 16) {
            s_z[wave][0][lane] = acc[0];   // batch 0
            s_z[wave][1][lane] = acc[1];   // batch 1
        }
    }
    __syncthreads();

    // ---- phase 4: gate fusion for the 16 columns, both batches ----
    if (tid < 32) {
        int b   = tid >> 4;
        int col = tid & 15;
        int h   = blockIdx.x * 16 + col;
        float zf = s_z[0][b][col] + bfv[h];
        float zi = s_z[1][b][col] + biv[h];
        float zg = s_z[2][b][col] + bgv[h];
        float zo = s_z[3][b][col] + bov[h];
        float zp = s_z[4][b][col] + bpv[h];
        size_t pidx = ((size_t)b * Dn + d) * Hn + h;
        float pf = pF[pidx], pi = pI[pidx], po = pO[pidx];

        float f = sigm(clip8(zf) + pf);
        float i = sigm(clip8(zi) + pi);
        float g = clip8(gelu(zg) + pi);
        float o = sigm(clip8(zo) + po);

        float cold = c_in[b * Hn + h];
        float hold = h_in[b * Hn + h];
        float c2 = cold * f + i * g;
        float h2 = o * clip8(gelu(zp));

        bool keep = mask[b * Dn + d] != 0;
        c2 = keep ? c2 : cold;
        h2 = keep ? h2 : hold;
        c_out[b * Hn + h] = clip10(c2);
        h_out[b * Hn + h] = clip10(h2);
    }
}

extern "C" void kernel_launch(void* const* d_in, const int* in_sizes, int n_in,
                              void* d_out, int out_size, void* d_ws, size_t ws_size,
                              hipStream_t stream) {
    (void)in_sizes; (void)n_in; (void)out_size; (void)ws_size;
    const float*         trans = (const float*)d_in[0];
    const unsigned char* mask  = (const unsigned char*)d_in[1];
    const float* Wf = (const float*)d_in[2];
    const float* bf = (const float*)d_in[3];
    const float* Wi = (const float*)d_in[4];
    const float* bi = (const float*)d_in[5];
    const float* Wg = (const float*)d_in[6];
    const float* bg = (const float*)d_in[7];
    const float* Wo = (const float*)d_in[8];
    const float* bo = (const float*)d_in[9];
    const float* Wp = (const float*)d_in[10];
    const float* bp = (const float*)d_in[11];
    const float* tF = (const float*)d_in[12];
    const float* tI = (const float*)d_in[13];
    const float* tO = (const float*)d_in[14];
    const float* cg = (const float*)d_in[15];
    const float* cb = (const float*)d_in[16];
    const float* hg = (const float*)d_in[17];
    const float* hb = (const float*)d_in[18];

    char* ws = (char*)d_ws;
    unsigned short* Wb = (unsigned short*)ws;                  // 5*H*H bf16 = 10 MB
    float* pF = (float*)(ws + (size_t)5 * Hn * Hn * 2);        // 3 * B*D*H fp32
    float* pI = pF + (size_t)Bn * Dn * Hn;
    float* pO = pI + (size_t)Bn * Dn * Hn;
    float* st = pO + (size_t)Bn * Dn * Hn;                     // 4 state buffers
    float* cA = st;
    float* hA = cA + Bn * Hn;
    float* cB = hA + Bn * Hn;
    float* hB = cB + Bn * Hn;

    // one-time prep (cheap; dominated by the 256 MB projection pass)
    w2bf_kernel<<<(5 * Hn * Hn) / (256 * 4), 256, 0, stream>>>(Wf, Wi, Wg, Wo, Wp, Wb);
    proj_kernel<<<Bn * Dn, 256, 0, stream>>>(trans, tF, tI, tO, pF, pI, pO);
    init_kernel<<<1, 256, 0, stream>>>(cA, hA);

    // 64 sequential recurrence steps, ping-pong state; last step writes d_out
    for (int d = 0; d < Dn; ++d) {
        const float* c_in = (d & 1) ? cB : cA;
        const float* h_in = (d & 1) ? hB : hA;
        float* c_out = (d & 1) ? cA : cB;
        float* h_out = (d == Dn - 1) ? (float*)d_out : ((d & 1) ? hA : hB);
        step_kernel<<<Hn / 16, 256, 0, stream>>>(Wb, pF, pI, pO,
                                                 bf, bi, bg, bo, bp,
                                                 cg, cb, hg, hb,
                                                 mask, c_in, h_in, c_out, h_out, d);
    }
}
